// GcnEncoderGraph_9002251452429
// MI455X (gfx1250) — matrix operand order repaired and checked by reading, not compiled, not run on hardware
//
#include <hip/hip_runtime.h>

// ---------------------------------------------------------------------------
// GCN encoder for MI455X (gfx1250, wave32, WMMA).
// B=8, N=2048, DIN=32, DH=DE=64, R=3.
// Strategy:
//   k_layer1 : y = sum_i (rel==i+1) @ x @ W1[i] + b ; row-L2-norm ; relu
//              also packs adjacency (rel!=0) into a bitmask for layers 2/3.
//   k_bn     : BatchNorm per node index over (batch, feature), in place.
//   k_max    : max over nodes -> d_out "output" region.
//   k_layer23: y = (adj @ h) @ W[0] + b ; row-L2-norm ; optional relu.
//   k_final  : ypred = output @ w_map + b_map.
// All big GEMMs use v_wmma_f32_16x16x32_f16; mask A-fragments are built in
// registers from relation ints / adjacency bits; B-fragments are staged in
// LDS transposed so each lane reads 16 contiguous K halves (2x ds_load_b128).
// ---------------------------------------------------------------------------

constexpr int kB   = 8;
constexpr int kN   = 2048;
constexpr int kDin = 32;
constexpr int kDh  = 64;

typedef __attribute__((ext_vector_type(16))) _Float16 v16h;
typedef __attribute__((ext_vector_type(8)))  _Float16 v8h;
typedef __attribute__((ext_vector_type(8)))  float    v8f;

static __device__ __forceinline__ v8f wmma_f16(v16h a, v16h b, v8f c) {
  // D = A(16x32 f16) * B(32x16 f16) + C(16x16 f32)
  return __builtin_amdgcn_wmma_f32_16x16x32_f16(
      /*neg_a=*/false, a, /*neg_b=*/false, b,
      /*c_mod=*/(short)0, c, /*reuse_a=*/false, /*reuse_b=*/false);
}

// ---------------------------------------------------------------------------
// Layer 1: per workgroup = 64 output rows of one batch (4 waves x 16 rows).
// ---------------------------------------------------------------------------
__global__ __launch_bounds__(128) void k_layer1(
    const float* __restrict__ x,        // [B,N,32]
    const int*   __restrict__ rel,      // [B,N,N]
    const float* __restrict__ w1,       // [3,32,64]
    const float* __restrict__ b1,       // [64]
    float* __restrict__ h1,             // [B,N,64]
    unsigned char* __restrict__ adjb)   // [B,N,N/8] packed bits
{
  __shared__ __attribute__((aligned(32))) _Float16 xT[kDin * 32];     // [col][k]
  __shared__ __attribute__((aligned(32))) _Float16 hs[4][3][16 * 32]; // per-wave h_i
  __shared__ float ys[4][16 * 64];                                    // per-wave y

  const int b    = blockIdx.y;
  const int wave = threadIdx.x >> 5;
  const int lane = threadIdx.x & 31;
  const int m    = lane & 15;       // row within 16-row tile
  const int hiH  = lane >> 4;       // lane group 0/1
  const int klo  = hiH ? 16 : 0;    // A-fragment K groups: contiguous k = 16h + s
  const int khi  = klo + 8;
  const int row0 = blockIdx.x * 64 + wave * 16;
  const int row  = row0 + m;

  const float* xb = x + (size_t)b * kN * kDin;
  const int* relrow = rel + ((size_t)b * kN + row) * kN;
  unsigned char* adjrow = adjb + ((size_t)b * kN + row) * (kN / 8);

  v8f acc[3][2] = {};

  for (int kc = 0; kc < kN; kc += 32) {
    __syncthreads();
    {  // stage x[kc:kc+32, 0:32] transposed to f16 LDS: xT[c][k]
      const int c  = threadIdx.x & 31;
      const int k0 = (threadIdx.x >> 5) * 8;
      v8h t;
#pragma unroll
      for (int j = 0; j < 8; ++j)
        t[j] = (_Float16)xb[(size_t)(kc + k0 + j) * kDin + c];
      *(v8h*)&xT[c * 32 + k0] = t;
    }
    __syncthreads();

    // relation ints for this lane's A-fragment slots (contiguous k = 16h + s order)
    int rv[16];
    const int* p = relrow + kc;
    *(int4*)&rv[0]  = *(const int4*)(p + klo);
    *(int4*)&rv[4]  = *(const int4*)(p + klo + 4);
    *(int4*)&rv[8]  = *(const int4*)(p + khi);
    *(int4*)&rv[12] = *(const int4*)(p + khi + 4);

    v16h a0, a1, a2;
    unsigned by0 = 0, by1 = 0;
#pragma unroll
    for (int e = 0; e < 8; ++e) {
      const int rl = rv[e], rh = rv[8 + e];
      a0[e]     = (_Float16)(rl == 1);
      a1[e]     = (_Float16)(rl == 2);
      a2[e]     = (_Float16)(rl == 3);
      a0[8 + e] = (_Float16)(rh == 1);
      a1[8 + e] = (_Float16)(rh == 2);
      a2[8 + e] = (_Float16)(rh == 3);
      by0 |= (unsigned)(rl != 0) << e;
      by1 |= (unsigned)(rh != 0) << e;
    }
    // pack adjacency bits (each lane owns 2 disjoint bytes of the 32-bit word)
    adjrow[(kc + klo) >> 3] = (unsigned char)by0;
    adjrow[(kc + khi) >> 3] = (unsigned char)by1;

    const int kb = hiH * 16;  // B-fragment: lanes 0-15 K 0..15, lanes 16-31 K 16..31
#pragma unroll
    for (int f = 0; f < 2; ++f) {
      const v16h bf = *(const v16h*)&xT[(f * 16 + m) * 32 + kb];
      acc[0][f] = wmma_f16(a0, bf, acc[0][f]);
      acc[1][f] = wmma_f16(a1, bf, acc[1][f]);
      acc[2][f] = wmma_f16(a2, bf, acc[2][f]);
    }
  }

  // ---- epilogue: y[16,64] = sum_i h_i(16x32) @ W1[i](32x64), via WMMA ----
#pragma unroll
  for (int i = 0; i < 3; ++i)
#pragma unroll
    for (int f = 0; f < 2; ++f)
#pragma unroll
      for (int r = 0; r < 8; ++r)
        hs[wave][i][(r + hiH * 8) * 32 + f * 16 + m] = (_Float16)acc[i][f][r];

  v8f yacc[4] = {};
  const int kb = hiH * 16;
#pragma unroll
  for (int i = 0; i < 3; ++i) {
    const _Float16* hp = &hs[wave][i][0];
    const v8h lo = *(const v8h*)&hp[m * 32 + klo];
    const v8h hi = *(const v8h*)&hp[m * 32 + khi];
    v16h ah;
#pragma unroll
    for (int e = 0; e < 8; ++e) { ah[e] = lo[e]; ah[8 + e] = hi[e]; }
    const float* wp = w1 + (size_t)i * kDin * kDh;
#pragma unroll
    for (int f = 0; f < 4; ++f) {
      const int c = f * 16 + m;
      v16h wb;
#pragma unroll
      for (int e = 0; e < 16; ++e)
        wb[e] = (_Float16)wp[(size_t)(kb + e) * kDh + c];
      yacc[f] = wmma_f16(ah, wb, yacc[f]);
    }
  }

  // bias -> LDS, then per-row L2 norm + relu + store
  float* yw = &ys[wave][0];
#pragma unroll
  for (int f = 0; f < 4; ++f) {
    const int c = f * 16 + m;
    const float bias = b1[c];
#pragma unroll
    for (int r = 0; r < 8; ++r)
      yw[(r + hiH * 8) * 64 + c] = yacc[f][r] + bias;
  }

  float ss = 0.f;
#pragma unroll
  for (int c = 0; c < 32; ++c) {
    const float v = yw[m * 64 + hiH * 32 + c];
    ss += v * v;
  }
  ss += __shfl_xor(ss, 16, 32);
  const float scale = 1.0f / fmaxf(sqrtf(ss), 1e-12f);

  float* orow = h1 + ((size_t)b * kN + row) * kDh + hiH * 32;
#pragma unroll
  for (int c = 0; c < 32; c += 4) {
    float4 v4;
    v4.x = fmaxf(yw[m * 64 + hiH * 32 + c + 0] * scale, 0.f);
    v4.y = fmaxf(yw[m * 64 + hiH * 32 + c + 1] * scale, 0.f);
    v4.z = fmaxf(yw[m * 64 + hiH * 32 + c + 2] * scale, 0.f);
    v4.w = fmaxf(yw[m * 64 + hiH * 32 + c + 3] * scale, 0.f);
    *(float4*)(orow + c) = v4;
  }
}

// ---------------------------------------------------------------------------
// Layers 2/3: y = (adj @ h)(16x64) @ W(64x64) + b ; norm ; optional relu.
// adj A-fragments are expanded from the packed bitmask (4 MB vs 128 MB).
// ---------------------------------------------------------------------------
__global__ __launch_bounds__(128) void k_layer23(
    const float* __restrict__ hin,            // [B,N,64]
    const unsigned char* __restrict__ adjb,   // [B,N,N/8]
    const float* __restrict__ W,              // [64,64] (relation 0 slice)
    const float* __restrict__ bias,           // [64]
    float* __restrict__ hout,                 // [B,N,64]
    const int do_relu)
{
  __shared__ __attribute__((aligned(32))) _Float16 hT[kDh * 32];      // [col][k]
  __shared__ __attribute__((aligned(32))) _Float16 ags[4][16 * 64];   // per-wave agg
  __shared__ float ys[4][16 * 64];

  const int b    = blockIdx.y;
  const int wave = threadIdx.x >> 5;
  const int lane = threadIdx.x & 31;
  const int m    = lane & 15;
  const int hiH  = lane >> 4;
  const int klo  = hiH ? 16 : 0;
  const int khi  = klo + 8;
  const int row0 = blockIdx.x * 64 + wave * 16;
  const int row  = row0 + m;

  const float* hb = hin + (size_t)b * kN * kDh;
  const unsigned char* adjrow = adjb + ((size_t)b * kN + row) * (kN / 8);

  v8f acc[4] = {};

  for (int kc = 0; kc < kN; kc += 32) {
    __syncthreads();
    {  // stage h[kc:kc+32, 0:64] transposed to f16 LDS
      const int c  = threadIdx.x & 63;
      const int k0 = (threadIdx.x >> 6) * 16;
      v8h t0, t1;
#pragma unroll
      for (int j = 0; j < 8; ++j) {
        t0[j] = (_Float16)hb[(size_t)(kc + k0 + j) * kDh + c];
        t1[j] = (_Float16)hb[(size_t)(kc + k0 + 8 + j) * kDh + c];
      }
      *(v8h*)&hT[c * 32 + k0]     = t0;
      *(v8h*)&hT[c * 32 + k0 + 8] = t1;
    }
    __syncthreads();

    const unsigned w32 = *(const unsigned*)(adjrow + (kc >> 3));
    v16h aa;
#pragma unroll
    for (int e = 0; e < 8; ++e) {
      aa[e]     = (_Float16)((w32 >> (klo + e)) & 1u);
      aa[8 + e] = (_Float16)((w32 >> (khi + e)) & 1u);
    }
    const int kb = hiH * 16;
#pragma unroll
    for (int f = 0; f < 4; ++f) {
      const v16h bf = *(const v16h*)&hT[(f * 16 + m) * 32 + kb];
      acc[f] = wmma_f16(aa, bf, acc[f]);
    }
  }

  // ---- epilogue: y = agg(16x64) @ W(64x64), K split in two 32-chunks ----
  _Float16* ag = &ags[wave][0];
#pragma unroll
  for (int f = 0; f < 4; ++f)
#pragma unroll
    for (int r = 0; r < 8; ++r)
      ag[(r + hiH * 8) * 64 + f * 16 + m] = (_Float16)acc[f][r];

  v8f yacc[4] = {};
  const int kb = hiH * 16;
#pragma unroll
  for (int kk = 0; kk < 2; ++kk) {
    const v8h lo = *(const v8h*)&ag[m * 64 + kk * 32 + klo];
    const v8h hi = *(const v8h*)&ag[m * 64 + kk * 32 + khi];
    v16h ah;
#pragma unroll
    for (int e = 0; e < 8; ++e) { ah[e] = lo[e]; ah[8 + e] = hi[e]; }
#pragma unroll
    for (int f = 0; f < 4; ++f) {
      const int c = f * 16 + m;
      v16h wb;
#pragma unroll
      for (int e = 0; e < 16; ++e)
        wb[e] = (_Float16)W[(size_t)(kk * 32 + kb + e) * kDh + c];
      yacc[f] = wmma_f16(ah, wb, yacc[f]);
    }
  }

  float* yw = &ys[wave][0];
#pragma unroll
  for (int f = 0; f < 4; ++f) {
    const int c = f * 16 + m;
    const float bv = bias[c];
#pragma unroll
    for (int r = 0; r < 8; ++r)
      yw[(r + hiH * 8) * 64 + c] = yacc[f][r] + bv;
  }

  float ss = 0.f;
#pragma unroll
  for (int c = 0; c < 32; ++c) {
    const float v = yw[m * 64 + hiH * 32 + c];
    ss += v * v;
  }
  ss += __shfl_xor(ss, 16, 32);
  const float scale = 1.0f / fmaxf(sqrtf(ss), 1e-12f);

  float* orow = hout + ((size_t)b * kN + row) * kDh + hiH * 32;
#pragma unroll
  for (int c = 0; c < 32; c += 4) {
    float4 v4;
    v4.x = yw[m * 64 + hiH * 32 + c + 0] * scale;
    v4.y = yw[m * 64 + hiH * 32 + c + 1] * scale;
    v4.z = yw[m * 64 + hiH * 32 + c + 2] * scale;
    v4.w = yw[m * 64 + hiH * 32 + c + 3] * scale;
    if (do_relu) {
      v4.x = fmaxf(v4.x, 0.f); v4.y = fmaxf(v4.y, 0.f);
      v4.z = fmaxf(v4.z, 0.f); v4.w = fmaxf(v4.w, 0.f);
    }
    *(float4*)(orow + c) = v4;
  }
}

// ---------------------------------------------------------------------------
// BatchNorm1d(N) training mode, affine identity: per node n, stats over
// (batch, feature) = 8*64 = 512 values. One wave per node, in place.
// ---------------------------------------------------------------------------
__global__ __launch_bounds__(256) void k_bn(float* __restrict__ h) {
  const int wave = threadIdx.x >> 5;
  const int lane = threadIdx.x & 31;
  const int n = blockIdx.x * 8 + wave;

  float vals[16];
  float s = 0.f, ss = 0.f;
  int idx = 0;
#pragma unroll
  for (int bb = 0; bb < kB; ++bb)
#pragma unroll
    for (int half = 0; half < 2; ++half) {
      const float v = h[((size_t)bb * kN + n) * kDh + half * 32 + lane];
      vals[idx++] = v; s += v; ss += v * v;
    }
#pragma unroll
  for (int o = 16; o >= 1; o >>= 1) {
    s  += __shfl_xor(s,  o, 32);
    ss += __shfl_xor(ss, o, 32);
  }
  const float mean = s * (1.f / 512.f);
  const float var  = ss * (1.f / 512.f) - mean * mean;
  const float sc   = rsqrtf(var + 1e-5f);
  idx = 0;
#pragma unroll
  for (int bb = 0; bb < kB; ++bb)
#pragma unroll
    for (int half = 0; half < 2; ++half) {
      h[((size_t)bb * kN + n) * kDh + half * 32 + lane] = (vals[idx++] - mean) * sc;
    }
}

// ---------------------------------------------------------------------------
// Max over nodes: out[b, off + d] = max_n h[b,n,d]. One block per batch.
// ---------------------------------------------------------------------------
__global__ __launch_bounds__(256) void k_max(const float* __restrict__ h,
                                             float* __restrict__ out, int off) {
  __shared__ float part[4][64];
  const int b = blockIdx.x;
  const int d = threadIdx.x & 63;
  const int strip = threadIdx.x >> 6;
  float mv = -3.402823466e+38f;
  const int n0 = strip * 512;
  for (int n = n0; n < n0 + 512; ++n)
    mv = fmaxf(mv, h[((size_t)b * kN + n) * kDh + d]);
  part[strip][d] = mv;
  __syncthreads();
  if (strip == 0) {
    mv = fmaxf(fmaxf(part[0][d], part[1][d]), fmaxf(part[2][d], part[3][d]));
    out[(size_t)b * 192 + off + d] = mv;
  }
}

// ---------------------------------------------------------------------------
// ypred[b,e] = sum_k output[b,k] * w_map[k,e] + b_map[e]
// ---------------------------------------------------------------------------
__global__ __launch_bounds__(512) void k_final(const float* __restrict__ out,
                                               const float* __restrict__ wmap,
                                               const float* __restrict__ bmap,
                                               float* __restrict__ ypred) {
  const int bb = threadIdx.x >> 6;
  const int e  = threadIdx.x & 63;
  float s = bmap[e];
#pragma unroll 4
  for (int k = 0; k < 192; ++k)
    s += out[bb * 192 + k] * wmap[(size_t)k * 64 + e];
  ypred[bb * 64 + e] = s;
}

// ---------------------------------------------------------------------------
extern "C" void kernel_launch(void* const* d_in, const int* in_sizes, int n_in,
                              void* d_out, int out_size, void* d_ws, size_t ws_size,
                              hipStream_t stream) {
  (void)in_sizes; (void)n_in; (void)out_size; (void)ws_size;

  const float* x       = (const float*)d_in[0];
  const int*   rel     = (const int*)d_in[1];
  // d_in[2] (float adjacency) unused: recomputed as a packed bitmask.
  const float* w_first = (const float*)d_in[3];
  const float* b_first = (const float*)d_in[4];
  const float* w_block = (const float*)d_in[5];
  const float* b_block = (const float*)d_in[6];
  const float* w_last  = (const float*)d_in[7];
  const float* b_last  = (const float*)d_in[8];
  const float* w_map   = (const float*)d_in[9];
  const float* b_map   = (const float*)d_in[10];

  float* out = (float*)d_out;                 // [B,192] then [B,64]

  float* h1 = (float*)d_ws;                               // 4 MB
  float* h2 = h1 + (size_t)kB * kN * kDh;                 // 4 MB
  float* h3 = h2 + (size_t)kB * kN * kDh;                 // 4 MB
  unsigned char* adjbits = (unsigned char*)(h3 + (size_t)kB * kN * kDh);  // 4 MB

  const dim3 gGemm(kN / 64, kB);
  const dim3 bGemm(128);

  // layer 1 (+ adjacency packing)
  k_layer1<<<gGemm, bGemm, 0, stream>>>(x, rel, w_first, b_first, h1, adjbits);
  k_bn<<<kN / 8, 256, 0, stream>>>(h1);
  k_max<<<kB, 256, 0, stream>>>(h1, out, 0);

  // layer 2 (relation slice 0 of w_block)
  k_layer23<<<gGemm, bGemm, 0, stream>>>(h1, adjbits, w_block, b_block, h2, 1);
  k_bn<<<kN / 8, 256, 0, stream>>>(h2);
  k_max<<<kB, 256, 0, stream>>>(h2, out, 64);

  // layer 3 (relation slice 0 of w_last; no relu/bn)
  k_layer23<<<gGemm, bGemm, 0, stream>>>(h2, adjbits, w_last, b_last, h3, 0);
  k_max<<<kB, 256, 0, stream>>>(h3, out, 128);

  // final linear into ypred region
  k_final<<<1, 512, 0, stream>>>(out, w_map, b_map, out + (size_t)kB * 192);
}
